// TopicScaledDotProductAttention_26938034881002
// MI455X (gfx1250) — matrix operand order repaired and checked by reading, not compiled, not run on hardware
//
#include <hip/hip_runtime.h>

#define B_  4
#define H_  8
#define TW_ 20
#define L_  256
#define D_  64

typedef __attribute__((ext_vector_type(16))) __bf16 v16bf;
typedef __attribute__((ext_vector_type(8)))  __bf16 v8bf;
typedef __attribute__((ext_vector_type(8)))  float  v8f;

__device__ __forceinline__ unsigned short f32_to_bf16(float f) {
    unsigned int u = __float_as_uint(f);
    u += 0x7FFFu + ((u >> 16) & 1u);   // round-to-nearest-even
    return (unsigned short)(u >> 16);
}

// -------- kernel 1: scores[b,h,w,l] = dot(q[b,h,w,:], k[b,h,l,:]) / 8 --------
__global__ __launch_bounds__(256) void scores_kernel(
        const float* __restrict__ q, const float* __restrict__ k,
        float* __restrict__ sc) {
    int w = blockIdx.x % TW_;
    int h = (blockIdx.x / TW_) % H_;
    int b = blockIdx.x / (TW_ * H_);
    __shared__ float qs[D_];
    if (threadIdx.x < D_)
        qs[threadIdx.x] = q[(((size_t)b * H_ + h) * TW_ + w) * D_ + threadIdx.x];
    __syncthreads();
    int l = threadIdx.x;  // 0..255
    const float4* krow = (const float4*)(k + (((size_t)b * H_ + h) * L_ + l) * D_);
    const float4* qv   = (const float4*)qs;
    float acc = 0.f;
#pragma unroll
    for (int d4 = 0; d4 < D_ / 4; ++d4) {
        float4 kk = krow[d4], qq = qv[d4];
        acc += qq.x * kk.x + qq.y * kk.y + qq.z * kk.z + qq.w * kk.w;
    }
    sc[(((size_t)b * H_ + h) * TW_ + w) * L_ + l] = acc * 0.125f;  // 1/sqrt(64)
}

// -------- kernel 2: per (b,l,h): softmax(scores+bias) @ v  --------
__global__ __launch_bounds__(256) void attn_kernel(
        const float* __restrict__ v, const float* __restrict__ bias,
        const float* __restrict__ sc, float* __restrict__ out) {

    __shared__ __align__(16) unsigned short w_lds[32 * L_];    // [32][256] bf16 (rows 20..31 pad)
    __shared__ __align__(16) unsigned short vT_lds[D_ * L_];   // [64][256] bf16: vT[n][k]

    const int tid  = threadIdx.x;
    const int lane = tid & 31;
    const int wave = tid >> 5;          // 0..7

    const int h  = blockIdx.x % H_;
    const int lb = (blockIdx.x / H_) % L_;
    const int b  = blockIdx.x / (H_ * L_);

    // ---- stage V transposed into LDS as bf16 (thread t owns k-row t) ----
    {
        const int kk = tid;  // 0..255
        const float4* vrow = (const float4*)(v + (((size_t)b * H_ + h) * L_ + kk) * D_);
#pragma unroll
        for (int n4 = 0; n4 < D_ / 4; ++n4) {
            float4 x = vrow[n4];
            vT_lds[(n4 * 4 + 0) * L_ + kk] = f32_to_bf16(x.x);
            vT_lds[(n4 * 4 + 1) * L_ + kk] = f32_to_bf16(x.y);
            vT_lds[(n4 * 4 + 2) * L_ + kk] = f32_to_bf16(x.z);
            vT_lds[(n4 * 4 + 3) * L_ + kk] = f32_to_bf16(x.w);
        }
    }

    // ---- zero-pad rows 20..31 of W ----
    {
        unsigned int* p = (unsigned int*)&w_lds[TW_ * L_];
        for (int i = tid; i < (32 - TW_) * L_ / 2; i += 256) p[i] = 0u;
    }

    // ---- softmax rows: wave w handles rows w, w+8, w+16 ----
    for (int r = wave; r < TW_; r += 8) {
        const float* brow = bias + ((((size_t)b * L_ + lb) * H_ + h) * TW_ + r) * L_;
        const float* srow = sc   + (((size_t)b * H_ + h) * TW_ + r) * L_;
        float x[8];
        float mx = -3.4e38f;
#pragma unroll
        for (int i = 0; i < 8; ++i) {
            int l = lane + 32 * i;                 // coalesced 128B/wave
            x[i] = brow[l] + srow[l];
            mx = fmaxf(mx, x[i]);
        }
#pragma unroll
        for (int off = 16; off > 0; off >>= 1)
            mx = fmaxf(mx, __shfl_xor(mx, off, 32));
        float s = 0.f;
#pragma unroll
        for (int i = 0; i < 8; ++i) { x[i] = __expf(x[i] - mx); s += x[i]; }
#pragma unroll
        for (int off = 16; off > 0; off >>= 1)
            s += __shfl_xor(s, off, 32);
        float inv = 1.0f / s;
#pragma unroll
        for (int i = 0; i < 8; ++i)
            w_lds[r * L_ + lane + 32 * i] = f32_to_bf16(x[i] * inv);
    }

    __syncthreads();

    // ---- WMMA: W[32(20),256]bf16 x Vt -> D[32(20),64]f32; wave owns one 16x16 tile ----
    const int mtile = wave >> 2;   // 0..1
    const int ntile = wave & 3;    // 0..3
    const int lh    = lane >> 4;   // lane-half selects K sub-ranges per ISA layout
    const int ll    = lane & 15;

    const unsigned short* arow = &w_lds[(mtile * 16 + ll) * L_];   // A row m
    const unsigned short* bcol = &vT_lds[(ntile * 16 + ll) * L_];  // B col n (as vT row)

    v8f acc = {};
#pragma unroll
    for (int kb = 0; kb < L_; kb += 32) {
        union { v16bf v; v8bf h[2]; } A, Bm;
        // A 16x32 bf16: lanes 0-15 -> K {0..7,16..23}; lanes 16-31 -> K {8..15,24..31}
        A.h[0]  = *(const v8bf*)(arow + kb + lh * 8);
        A.h[1]  = *(const v8bf*)(arow + kb + lh * 8 + 16);
        // B 32x16 bf16: lanes 0-15 -> K {0..7,16..23}; lanes 16-31 -> K {8..15,24..31}
        Bm.h[0] = *(const v8bf*)(bcol + kb + lh * 8);
        Bm.h[1] = *(const v8bf*)(bcol + kb + lh * 8 + 16);
        acc = __builtin_amdgcn_wmma_f32_16x16x32_bf16(
            false, A.v, false, Bm.v, (short)0, acc, false, false);
    }

    // ---- store: C/D layout VGPR r -> (M = r + 8*lane_half, N = lane&15) ----
    const size_t obase = (((size_t)b * L_ + lb) * TW_) * (H_ * D_)
                       + (size_t)h * D_ + ntile * 16 + ll;
#pragma unroll
    for (int r = 0; r < 8; ++r) {
        int mg = mtile * 16 + lh * 8 + r;
        if (mg < TW_)
            out[obase + (size_t)mg * (H_ * D_)] = acc[r];
    }
}

extern "C" void kernel_launch(void* const* d_in, const int* in_sizes, int n_in,
                              void* d_out, int out_size, void* d_ws, size_t ws_size,
                              hipStream_t stream) {
    const float* q    = (const float*)d_in[0];
    const float* k    = (const float*)d_in[1];
    const float* v    = (const float*)d_in[2];
    const float* bias = (const float*)d_in[3];
    float* out = (float*)d_out;
    float* sc  = (float*)d_ws;   // B*H*TW*L fp32 = 640 KB scratch

    scores_kernel<<<B_ * H_ * TW_, 256, 0, stream>>>(q, k, sc);
    attn_kernel<<<B_ * L_ * H_, 256, 0, stream>>>(v, bias, sc, out);
}
